// SimplifiedSplatFlowAttention_43224550867618
// MI455X (gfx1250) — matrix operand order repaired and checked by reading, not compiled, not run on hardware
//
#include <hip/hip_runtime.h>

// ---------------------------------------------------------------------------
// Types for CDNA5 WMMA (wave32): v_wmma_f32_16x16x32_bf16
// ---------------------------------------------------------------------------
typedef __attribute__((ext_vector_type(16))) __bf16 v16bf;
typedef __attribute__((ext_vector_type(8)))  float  v8f;

union FragBF16 {
    uint4 q[2];   // 32 bytes
    v16bf v;
};
static_assert(sizeof(FragBF16) == 32, "frag size");

__device__ __forceinline__ unsigned short f2bf(float f) {
    unsigned int u = __float_as_uint(f);
    unsigned int r = u + 0x7FFFu + ((u >> 16) & 1u);   // round-to-nearest-even
    return (unsigned short)(r >> 16);
}
__device__ __forceinline__ float bf2f(unsigned short h) {
    return __uint_as_float(((unsigned int)h) << 16);
}

#define TOKENS 32768   // B*S
#define DIM    1024
#define NSPLAT 12
#define SEQ    8192
#define NBATCH 4

// ---------------------------------------------------------------------------
// K0: prep — cast weights to bf16, per-splat constants, zero splat accumulator
// grid = 2048 (weight cast) + 1 (cparams) + 256 (zero sv) = 2305 blocks x 256
// ---------------------------------------------------------------------------
__global__ __launch_bounds__(256) void k0_prep(
    const float* __restrict__ wv, const float* __restrict__ wo,
    const float* __restrict__ centers, const float* __restrict__ scales,
    const float* __restrict__ amps,
    unsigned short* __restrict__ wvb, unsigned short* __restrict__ wob,
    float* __restrict__ cparams, float* __restrict__ sv)
{
    int b = blockIdx.x, t = threadIdx.x;
    if (b < 2048) {
        int which = (b >= 1024);
        int idx = (which ? (b - 1024) : b) * 1024 + t * 4;
        const float* src = which ? wo : wv;
        unsigned short* dst = which ? wob : wvb;
        float4 v = *(const float4*)&src[idx];
        ushort4 o;
        o.x = f2bf(v.x); o.y = f2bf(v.y); o.z = f2bf(v.z); o.w = f2bf(v.w);
        *(ushort4*)&dst[idx] = o;
    } else if (b == 2048) {
        int wave = t >> 5, lane = t & 31;
        for (int n = wave; n < NSPLAT; n += 8) {
            float a = 0.f;
            for (int d = lane; d < DIM; d += 32) {
                float c = centers[n * DIM + d];
                a = fmaf(c, c, a);
            }
            #pragma unroll
            for (int off = 16; off >= 1; off >>= 1)
                a += __shfl_xor(a, off, 32);
            if (lane == 0) {
                cparams[n] = a;                                  // c2
                float sc = __expf(scales[n]);
                sc = fminf(fmaxf(sc, 0.1f), 2.0f);
                cparams[16 + n] = 0.5f / (sc * sc);              // 0.5/sigma^2
                cparams[32 + n] = 1.f / (1.f + __expf(-amps[n])); // sigmoid(amp)
            }
        }
    } else {
        int idx = (b - 2049) * 256 + t;   // 256 blocks * 256 = 65536 floats
        sv[idx] = 0.f;
    }
}

// ---------------------------------------------------------------------------
// K1: splat weights (one wave per token) + bf16 copy of x
// grid = 4096 blocks x 256 (8 waves = 8 tokens per block)
// ---------------------------------------------------------------------------
__global__ __launch_bounds__(256) void k1_weights(
    const float* __restrict__ x, const float* __restrict__ centers,
    const float* __restrict__ cparams,
    unsigned short* __restrict__ xb, float* __restrict__ wgt)
{
    __shared__ float cl[NSPLAT * DIM];   // 48 KB
    int t = threadIdx.x;
    for (int i = t * 4; i < NSPLAT * DIM; i += 256 * 4)
        *(float4*)&cl[i] = *(const float4*)&centers[i];
    __syncthreads();

    int wave = t >> 5, lane = t & 31;
    long token = (long)blockIdx.x * 8 + wave;
    const float* xp = x + token * DIM;
    unsigned short* xbp = xb + token * DIM;

    float acc[NSPLAT];
    #pragma unroll
    for (int n = 0; n < NSPLAT; ++n) acc[n] = 0.f;
    float x2 = 0.f;

    #pragma unroll
    for (int i = 0; i < 8; ++i) {
        int d = i * 128 + lane * 4;
        float4 xv = *(const float4*)&xp[d];
        ushort4 o;
        o.x = f2bf(xv.x); o.y = f2bf(xv.y); o.z = f2bf(xv.z); o.w = f2bf(xv.w);
        *(ushort4*)&xbp[d] = o;
        x2 = fmaf(xv.x, xv.x, fmaf(xv.y, xv.y, fmaf(xv.z, xv.z, fmaf(xv.w, xv.w, x2))));
        #pragma unroll
        for (int n = 0; n < NSPLAT; ++n) {
            float4 cv = *(const float4*)&cl[n * DIM + d];
            acc[n] = fmaf(xv.x, cv.x, fmaf(xv.y, cv.y, fmaf(xv.z, cv.z, fmaf(xv.w, cv.w, acc[n]))));
        }
    }

    #pragma unroll
    for (int off = 16; off >= 1; off >>= 1) {
        x2 += __shfl_xor(x2, off, 32);
        #pragma unroll
        for (int n = 0; n < NSPLAT; ++n)
            acc[n] += __shfl_xor(acc[n], off, 32);
    }

    float wv[NSPLAT], s = 0.f;
    #pragma unroll
    for (int n = 0; n < NSPLAT; ++n) {
        float d2 = fmaxf(x2 + cparams[n] - 2.f * acc[n], 0.f);
        float g  = __expf(-d2 * cparams[16 + n]) * cparams[32 + n];
        wv[n] = g; s += g;
    }
    float inv = 1.f / fmaxf(s, 1e-8f);
    if (lane == 0) {
        #pragma unroll
        for (int n = 0; n < 16; ++n)
            wgt[token * 16 + n] = (n < NSPLAT) ? wv[n] * inv : 0.f;
    }
}

// ---------------------------------------------------------------------------
// K2/K5: C[m,n] = sum_k A[m,k] * Bw[n,k]   (A,Bw bf16 row-major, K=N=1024)
// Block tile 256x128, k-step 32, 8 waves; each wave owns 32 M-rows x 128 N:
// per k-step = 2 A-frags + 8 B-frags feeding 16 v_wmma_f32_16x16x32_bf16
// (each B frag reused by 2 back-to-back WMMAs -> 1.25 ds_load per wmma).
// LDS tiles padded to stride 40 (bank-conflict-free b128 reads).
// grid = (M/256, 1024/128) = (128, 8), block = 256.
// ---------------------------------------------------------------------------
__global__ __launch_bounds__(256) void k_gemm_bf16(
    const unsigned short* __restrict__ A,
    const unsigned short* __restrict__ Bw,
    void* __restrict__ Cout, int storeF32)
{
    __shared__ __align__(16) unsigned short Asl[256 * 40];  // 20 KB
    __shared__ __align__(16) unsigned short Bsl[128 * 40];  // 10 KB

    const int K = DIM;
    int m0 = blockIdx.x * 256;
    int n0 = blockIdx.y * 128;
    int t = threadIdx.x;
    int wave = t >> 5, lane = t & 31;
    int h = lane >> 4, l16 = lane & 15;
    int lrow = t >> 1, lhalf = t & 1;

    v8f acc[2][8];
    #pragma unroll
    for (int u = 0; u < 2; ++u)
        #pragma unroll
        for (int j = 0; j < 8; ++j)
            #pragma unroll
            for (int r = 0; r < 8; ++r) acc[u][j][r] = 0.f;

    for (int k0 = 0; k0 < K; k0 += 32) {
        __syncthreads();
        {   // A tile: 256 rows x 32 k (two 16-bf16 chunks per thread)
            const uint4* g0 = (const uint4*)(A + (size_t)(m0 + lrow) * K + k0 + lhalf * 16);
            const uint4* g1 = (const uint4*)(A + (size_t)(m0 + 128 + lrow) * K + k0 + lhalf * 16);
            uint4* d0 = (uint4*)&Asl[lrow * 40 + lhalf * 16];
            uint4* d1 = (uint4*)&Asl[(128 + lrow) * 40 + lhalf * 16];
            d0[0] = g0[0]; d0[1] = g0[1];
            d1[0] = g1[0]; d1[1] = g1[1];
        }
        {   // B tile: rows n0..n0+127 of Bw, 32 k — stored [n][k]
            const uint4* g = (const uint4*)(Bw + (size_t)(n0 + lrow) * K + k0 + lhalf * 16);
            uint4* d = (uint4*)&Bsl[lrow * 40 + lhalf * 16];
            d[0] = g[0]; d[1] = g[1];
        }
        __syncthreads();

        // A fragments (16x32 bf16 each): lane half h -> comps 0..7 = k 8h..8h+7,
        // comps 8..15 = k 16+8h..23+8h  (ISA 7.12.2 16-bit A layout)
        FragBF16 a0, a1;
        {
            const unsigned short* r0 = &Asl[(wave * 32 + l16) * 40];
            const unsigned short* r1 = &Asl[(wave * 32 + 16 + l16) * 40];
            a0.q[0] = *(const uint4*)&r0[8 * h];
            a0.q[1] = *(const uint4*)&r0[16 + 8 * h];
            a1.q[0] = *(const uint4*)&r1[8 * h];
            a1.q[1] = *(const uint4*)&r1[16 + 8 * h];
        }

        #pragma unroll
        for (int j = 0; j < 8; ++j) {
            // B fragment (32x16 bf16): lane half h -> comps 0..7 = k 8h..8h+7, comps 8..15 = k 16+8h..23+8h at col l16
            FragBF16 bf;
            const unsigned short* row = &Bsl[(j * 16 + l16) * 40];
            bf.q[0] = *(const uint4*)&row[8 * h];
            bf.q[1] = *(const uint4*)&row[16 + 8 * h];
            acc[0][j] = __builtin_amdgcn_wmma_f32_16x16x32_bf16(
                false, a0.v, false, bf.v, (short)0, acc[0][j], false, false);
            acc[1][j] = __builtin_amdgcn_wmma_f32_16x16x32_bf16(
                false, a1.v, false, bf.v, (short)0, acc[1][j], false, false);
        }
    }

    // C layout: lanes 0-15 VGPR r -> M=r; lanes 16-31 -> M=8+r; N=l16
    #pragma unroll
    for (int u = 0; u < 2; ++u) {
        #pragma unroll
        for (int j = 0; j < 8; ++j) {
            int n = n0 + j * 16 + l16;
            #pragma unroll
            for (int r = 0; r < 8; ++r) {
                int m = m0 + wave * 32 + u * 16 + h * 8 + r;
                float v = acc[u][j][r];
                if (storeF32)
                    ((float*)Cout)[(size_t)m * DIM + n] = v;
                else
                    ((unsigned short*)Cout)[(size_t)m * DIM + n] = f2bf(v);
            }
        }
    }
}

// ---------------------------------------------------------------------------
// K3: sv[b,n,d] += sum_s w[b,s,n] * values[b,s,d]
// grid = (4 d-chunks, 16 s-chunks, 4 batches), block = 256; w chunk in LDS.
// ---------------------------------------------------------------------------
__global__ __launch_bounds__(256) void k3_reduce(
    const unsigned short* __restrict__ values, const float* __restrict__ wgt,
    float* __restrict__ sv)
{
    __shared__ float wl[512 * 16];  // 32 KB
    int b = blockIdx.z;
    int d = blockIdx.x * 256 + threadIdx.x;
    long base = (long)b * SEQ + (long)blockIdx.y * 512;

    for (int i = threadIdx.x * 4; i < 512 * 16; i += 1024)
        *(float4*)&wl[i] = *(const float4*)&wgt[base * 16 + i];
    __syncthreads();

    float acc[NSPLAT];
    #pragma unroll
    for (int n = 0; n < NSPLAT; ++n) acc[n] = 0.f;

    const unsigned short* vp = values + base * DIM + d;
    for (int s = 0; s < 512; ++s) {
        float v = bf2f(vp[(long)s * DIM]);
        const float* wr = &wl[s * 16];
        #pragma unroll
        for (int n = 0; n < NSPLAT; ++n)
            acc[n] = fmaf(wr[n], v, acc[n]);
    }
    float* svp = sv + (long)b * 16 * DIM + d;
    #pragma unroll
    for (int n = 0; n < NSPLAT; ++n)
        atomicAdd(&svp[n * DIM], acc[n]);
}

// ---------------------------------------------------------------------------
// K4: t[m,d] = sum_n w[m,n] * sv[b,n,d]  -> bf16 temp (reuses xb region)
// grid = 512 blocks x 256 (64 tokens per block, same batch), sv in LDS.
// ---------------------------------------------------------------------------
__global__ __launch_bounds__(256) void k4_broadcast(
    const float* __restrict__ wgt, const float* __restrict__ sv,
    unsigned short* __restrict__ tb)
{
    __shared__ float svl[NSPLAT * DIM];   // 48 KB
    long token0 = (long)blockIdx.x * 64;
    int b = (int)(token0 / SEQ);
    const float* svp = sv + (long)b * 16 * DIM;   // rows n=0..11 contiguous
    for (int i = threadIdx.x * 4; i < NSPLAT * DIM; i += 1024)
        *(float4*)&svl[i] = *(const float4*)&svp[i];
    __syncthreads();

    int wave = threadIdx.x >> 5, lane = threadIdx.x & 31;
    for (int it = 0; it < 8; ++it) {
        long token = token0 + wave * 8 + it;
        const float* wp = wgt + token * 16;
        float wr[NSPLAT];
        {
            float4 a = *(const float4*)&wp[0];
            float4 c = *(const float4*)&wp[4];
            float4 e = *(const float4*)&wp[8];
            wr[0]=a.x; wr[1]=a.y; wr[2]=a.z;  wr[3]=a.w;
            wr[4]=c.x; wr[5]=c.y; wr[6]=c.z;  wr[7]=c.w;
            wr[8]=e.x; wr[9]=e.y; wr[10]=e.z; wr[11]=e.w;
        }
        unsigned short* op = tb + token * DIM;
        #pragma unroll
        for (int i = 0; i < 8; ++i) {
            int d = i * 128 + lane * 4;
            float4 r = make_float4(0.f, 0.f, 0.f, 0.f);
            #pragma unroll
            for (int n = 0; n < NSPLAT; ++n) {
                float4 cv = *(const float4*)&svl[n * DIM + d];
                r.x = fmaf(wr[n], cv.x, r.x);
                r.y = fmaf(wr[n], cv.y, r.y);
                r.z = fmaf(wr[n], cv.z, r.z);
                r.w = fmaf(wr[n], cv.w, r.w);
            }
            ushort4 o;
            o.x = f2bf(r.x); o.y = f2bf(r.y); o.z = f2bf(r.z); o.w = f2bf(r.w);
            *(ushort4*)&op[d] = o;
        }
    }
}

// ---------------------------------------------------------------------------
// Launch
// ---------------------------------------------------------------------------
extern "C" void kernel_launch(void* const* d_in, const int* in_sizes, int n_in,
                              void* d_out, int out_size, void* d_ws, size_t ws_size,
                              hipStream_t stream) {
    (void)in_sizes; (void)n_in; (void)out_size; (void)ws_size;
    const float* x       = (const float*)d_in[0];
    const float* centers = (const float*)d_in[1];
    const float* scales  = (const float*)d_in[2];
    const float* amps    = (const float*)d_in[3];
    const float* wv      = (const float*)d_in[4];
    const float* wo      = (const float*)d_in[5];

    char* ws = (char*)d_ws;
    const size_t off_xb  = 0;                          // 64 MiB bf16 x (reused as t-buf)
    const size_t off_val = (size_t)64 << 20;           // 64 MiB bf16 values
    const size_t off_wvb = (size_t)128 << 20;          // 2 MiB
    const size_t off_wob = off_wvb + ((size_t)2 << 20);
    const size_t off_wgt = off_wob + ((size_t)2 << 20); // [TOKENS,16] f32 = 2 MiB
    const size_t off_sv  = off_wgt + ((size_t)2 << 20); // [4,16,1024] f32 = 256 KiB
    const size_t off_cp  = off_sv + (size_t)64 * 1024 * 4;

    unsigned short* xb  = (unsigned short*)(ws + off_xb);
    unsigned short* val = (unsigned short*)(ws + off_val);
    unsigned short* wvb = (unsigned short*)(ws + off_wvb);
    unsigned short* wob = (unsigned short*)(ws + off_wob);
    float* wgt = (float*)(ws + off_wgt);
    float* sv  = (float*)(ws + off_sv);
    float* cp  = (float*)(ws + off_cp);

    k0_prep<<<2305, 256, 0, stream>>>(wv, wo, centers, scales, amps, wvb, wob, cp, sv);
    k1_weights<<<TOKENS / 8, 256, 0, stream>>>(x, centers, cp, xb, wgt);
    k_gemm_bf16<<<dim3(TOKENS / 256, DIM / 128), 256, 0, stream>>>(xb, wvb, (void*)val, 0);
    k3_reduce<<<dim3(4, 16, 4), 256, 0, stream>>>(val, wgt, sv);
    k4_broadcast<<<TOKENS / 64, 256, 0, stream>>>(wgt, sv, xb);
    k_gemm_bf16<<<dim3(TOKENS / 256, DIM / 128), 256, 0, stream>>>(xb, wob, d_out, 1);
}
